// GlobalGraph_43155831390242
// MI455X (gfx1250) — matrix operand order repaired and checked by reading, not compiled, not run on hardware
//
#include <hip/hip_runtime.h>

// Masked self-attention for B=64, N=1024, D_IN=D=128 on gfx1250 (MI455X).
// Flash-attention structure: scores never hit HBM (2x268MB vs ~70MB minimum
// traffic at 23.3 TB/s). bf16 WMMA 16x16x32 w/ f32 accumulation for matmuls.
// K/V^T tiles are DMA'd into LDS by the Tensor Data Mover, double-buffered
// (128KB LDS) so tile kb+1's DMA overlaps tile kb's WMMA/softmax compute;
// the wave only waits s_wait_tensorcnt<=2 (in-order completion) per tile.

#define BATCH 64
#define SEQ   1024
#define DIM   128
#define SCALE 0.08838834764831845f   // 1/sqrt(128)

typedef __attribute__((ext_vector_type(16))) __bf16 bf16x16;
typedef __attribute__((ext_vector_type(8)))  __bf16 bf16x8;
typedef __attribute__((ext_vector_type(8)))  float  f32x8;
typedef __attribute__((ext_vector_type(4)))  float  f32x4;
typedef __attribute__((ext_vector_type(4)))  unsigned int su32x4;
typedef __attribute__((ext_vector_type(8)))  unsigned int su32x8;

__device__ __forceinline__ f32x8 wmma_bf16(bf16x16 a, bf16x16 b, f32x8 c) {
    // D = A(16x32 bf16) * B(32x16 bf16) + C(16x16 f32)
    return __builtin_amdgcn_wmma_f32_16x16x32_bf16(false, a, false, b,
                                                   (short)0, c, false, false);
}

// A-fragment (16x32 bf16) load. Layout used here (same k order as B-frags): lane = row (lane&15);
// elems 0..15 -> k = c*32 + half*16 + e (contiguous), half = lane>>4.
__device__ __forceinline__ bf16x16 load_a_frag(const __bf16* rowptr, int c, int half) {
    bf16x8 lo = *(const bf16x8*)(rowptr + c * 32 + half * 16);
    bf16x8 hi = *(const bf16x8*)(rowptr + c * 32 + half * 16 + 8);
    return __builtin_shufflevector(lo, hi, 0, 1, 2, 3, 4, 5, 6, 7,
                                   8, 9, 10, 11, 12, 13, 14, 15);
}

// TDM: DMA a 2D bf16 tile (tile_d1 rows x tile_d0 elems, row stride stride0
// elems) from global memory into LDS at byte offset lds_addr.
// D# packing per CDNA5 ISA 8.3/8.4:
//  group0: [count=1][lds_addr][global_addr 56:0][type=2]
//  group1: [mask=0, data_size=2B][tensor dims][tile dims][dim0 stride]
//  groups2/3: zero (tile_dim2 = 0 -> dims 2+ unused)
__device__ __forceinline__ void tdm_load_tile_bf16(
    unsigned lds_addr, const __bf16* gptr,
    unsigned tile_d0, unsigned tile_d1,
    unsigned tensor_d0, unsigned tensor_d1, unsigned stride0)
{
    unsigned long long ga = (unsigned long long)(uintptr_t)gptr;
    su32x4 g0 = { 1u,                                   // count=1, user desc
                  lds_addr,
                  (unsigned)ga,
                  ((unsigned)(ga >> 32) & 0x01FFFFFFu) | (2u << 30) }; // type=2
    su32x8 g1 = { 0x00010000u,                          // data_size = 2 bytes
                  (tensor_d0 & 0xFFFFu) << 16,          // tensor_dim0 lo
                  (tensor_d0 >> 16) | ((tensor_d1 & 0xFFFFu) << 16),
                  (tensor_d1 >> 16) | (tile_d0 << 16),  // tile_dim0
                  tile_d1,                              // tile_dim1, tile_dim2=0
                  stride0,                              // tensor_dim0_stride lo
                  0u, 0u };
    su32x4 gz = { 0u, 0u, 0u, 0u };
    asm volatile("tensor_load_to_lds %0, %1, %2, %3"
                 :: "s"(g0), "s"(g1), "s"(gz), "s"(gz)
                 : "memory");
}

// Issue the {K, V^T} tile pair for key-block kb of batch b into the LDS
// double-buffer at byte offset bufbase (K at +0, V^T at +32KB).
__device__ __forceinline__ void issue_kv_tdm(
    const __bf16* kg, const __bf16* vtg, int b, int kb, unsigned bufbase)
{
    tdm_load_tile_bf16(bufbase,
        kg + ((size_t)b * SEQ + (size_t)kb * 128) * DIM,
        /*tile_d0*/DIM, /*tile_d1*/128,
        /*tensor_d0*/DIM, /*tensor_d1*/128, /*stride0*/DIM);
    tdm_load_tile_bf16(bufbase + 32768u,
        vtg + (size_t)b * DIM * SEQ + (size_t)kb * 128,
        /*tile_d0*/128, /*tile_d1*/DIM,
        /*tensor_d0*/128, /*tensor_d1*/DIM, /*stride0*/SEQ);
}

// ---------------------------------------------------------------------------
// Kernel 1: QKV projection. Q,K stored row-major bf16 [B*N][D]; V stored
// transposed bf16 [B][D][N] so PV B-fragments are contiguous in LDS later.
// ---------------------------------------------------------------------------
__global__ __launch_bounds__(256) void qkv_proj_kernel(
    const float* __restrict__ x,
    const float* __restrict__ Wq, const float* __restrict__ bq,
    const float* __restrict__ Wk, const float* __restrict__ bk,
    const float* __restrict__ Wv, const float* __restrict__ bv,
    __bf16* __restrict__ qg, __bf16* __restrict__ kg, __bf16* __restrict__ vtg)
{
    __shared__ __bf16 Wt[DIM * DIM];            // 32 KB: W transposed, bf16

    const int tid  = threadIdx.x;
    const int wave = tid >> 5;
    const int lane = tid & 31;
    const int col  = lane & 15;
    const int half = lane >> 4;

    const int row0 = blockIdx.x * 128 + wave * 16;   // flat row (b*N + n) tile

    // x tile (16x128 f32) -> 4 A-fragments (bf16), reused for Q,K,V
    const float* xrow = x + (size_t)(row0 + col) * DIM;
    bf16x16 xa[4];
#pragma unroll
    for (int c = 0; c < 4; ++c) {
        f32x4 l0 = *(const f32x4*)(xrow + c * 32 + half * 16);
        f32x4 l1 = *(const f32x4*)(xrow + c * 32 + half * 16 + 4);
        f32x4 h0 = *(const f32x4*)(xrow + c * 32 + half * 16 + 8);
        f32x4 h1 = *(const f32x4*)(xrow + c * 32 + half * 16 + 12);
        bf16x16 a;
#pragma unroll
        for (int e = 0; e < 4; ++e) {
            a[e]      = (__bf16)l0[e];
            a[4 + e]  = (__bf16)l1[e];
            a[8 + e]  = (__bf16)h0[e];
            a[12 + e] = (__bf16)h1[e];
        }
        xa[c] = a;
    }

    const float* Ws[3] = {Wq, Wk, Wv};
    const float* bs[3] = {bq, bk, bv};

#pragma unroll
    for (int p = 0; p < 3; ++p) {
        __syncthreads();
        // Load + transpose W (f32 [din][dout]) into LDS bf16 Wt[dout][din]
        const float* W = Ws[p];
        for (int i = tid; i < DIM * DIM; i += 256) {
            const int din = i >> 7, dout = i & 127;
            Wt[dout * DIM + din] = (__bf16)W[i];
        }
        __syncthreads();

#pragma unroll
        for (int t = 0; t < 8; ++t) {
            f32x8 acc = (f32x8)(bs[p][t * 16 + col]);  // bias splat (per-col)
#pragma unroll
            for (int c = 0; c < 4; ++c) {
                // B-frag: lane=col, k = c*32 + half*16 + e  (contiguous 32B)
                bf16x16 bfrag = *(const bf16x16*)(Wt + (t * 16 + col) * DIM
                                                  + c * 32 + half * 16);
                acc = wmma_bf16(xa[c], bfrag, acc);
            }
            if (p == 0 || p == 1) {
                __bf16* dst = (p == 0) ? qg : kg;
#pragma unroll
                for (int r = 0; r < 8; ++r)     // C layout: VGPR r = row r+8*half
                    dst[(size_t)(row0 + r + 8 * half) * DIM + t * 16 + col] =
                        (__bf16)acc[r];
            } else {
#pragma unroll
                for (int r = 0; r < 8; ++r) {
                    const int rg = row0 + r + 8 * half;
                    const int bb = rg >> 10, nn = rg & 1023;
                    vtg[((size_t)bb * DIM + t * 16 + col) * SEQ + nn] =
                        (__bf16)acc[r];
                }
            }
        }
    }
}

// ---------------------------------------------------------------------------
// Kernel 2: flash attention. 1 WG (8 waves) = 128 query rows of one batch.
// LDS double-buffer (128KB): buf0 = [K | V^T] at 0/32KB, buf1 at 64/96KB.
// Wave 0 keeps the next tile pair in flight on the TDM while all 8 waves
// run WMMAs on the current one; s_wait_tensorcnt(2) retires the current
// pair (per-wave in-order completion). The current K region is recycled as
// the P (probability) buffer after the S-phase barrier.
// ---------------------------------------------------------------------------
__global__ __launch_bounds__(256) void attn_kernel(
    const __bf16* __restrict__ qg, const __bf16* __restrict__ kg,
    const __bf16* __restrict__ vtg, const int* __restrict__ valid_lens,
    float* __restrict__ out)
{
    __shared__ __bf16 smem[2 * 2 * 128 * DIM];  // 128 KB: 2 x {K, Vt}

    const int tid  = threadIdx.x;
    const int wave = tid >> 5;
    const int lane = tid & 31;
    const int col  = lane & 15;
    const int half = lane >> 4;

    const int b  = blockIdx.x >> 3;
    const int qb = blockIdx.x & 7;
    const int L  = valid_lens[b];
    const int qn = qb * 128 + wave * 16;        // wave's first query row in batch

    if (qb * 128 >= L) {                        // whole block masked -> zeros
        float* o = out + ((size_t)b * SEQ + qn) * DIM;
#pragma unroll
        for (int r = 0; r < 8; ++r)
#pragma unroll
            for (int t = 0; t < 8; ++t)
                o[(size_t)(r + 8 * half) * DIM + t * 16 + col] = 0.f;
        return;
    }

    const int nkb = (L + 127) >> 7;             // skip fully-masked KV tiles

    // Prologue: start DMA of tile 0 into buf0 before touching Q.
    if (wave == 0) issue_kv_tdm(kg, vtg, b, 0, 0u);

    // Q stripe -> 4 A-fragments, resident for the whole kernel
    const __bf16* qrow = qg + ((size_t)b * SEQ + qn + col) * DIM;
    bf16x16 qa[4];
#pragma unroll
    for (int c = 0; c < 4; ++c) qa[c] = load_a_frag(qrow, c, half);

    float mrow[8], lrow[8];
    f32x8 o[8];
#pragma unroll
    for (int r = 0; r < 8; ++r) { mrow[r] = -3.0e38f; lrow[r] = 0.f; }
#pragma unroll
    for (int t = 0; t < 8; ++t) o[t] = (f32x8)(0.f);

    for (int kb = 0; kb < nkb; ++kb) {
        __bf16* Ks  = smem + (kb & 1) * 32768;  // current buffer (elements)
        __bf16* Vts = Ks + 16384;

        __syncthreads();                        // all waves done with buf[nxt]
        if (wave == 0) {
            if (kb + 1 < nkb) {                 // keep next pair in flight
                issue_kv_tdm(kg, vtg, b, kb + 1, (kb & 1) ? 0u : 65536u);
                __builtin_amdgcn_s_wait_tensorcnt(2);   // tile kb has landed
            } else {
                __builtin_amdgcn_s_wait_tensorcnt(0);
            }
        }
        if (kb + 2 < nkb) {                     // warm L2 two tiles ahead
            const __bf16* nk = kg + ((size_t)b * SEQ + (kb + 2) * 128 + (tid >> 1)) * DIM;
            __builtin_prefetch(nk, 0, 1);       // -> global_prefetch_b8
        }
        __syncthreads();                        // publish buf[cur] to all waves

        // S = Q K^T * scale, masked on invalid key columns
        f32x8 st[8];
#pragma unroll
        for (int t = 0; t < 8; ++t) {
            f32x8 s = (f32x8)(0.f);
            const __bf16* kr = Ks + (t * 16 + col) * DIM;   // lane = key col
#pragma unroll
            for (int c = 0; c < 4; ++c) {
                bf16x16 bfrag = *(const bf16x16*)(kr + c * 32 + half * 16);
                s = wmma_bf16(qa[c], bfrag, s);
            }
            s = s * SCALE;
            const int j = kb * 128 + t * 16 + col;
            st[t] = (j < L) ? s : (f32x8)(-1.0e12f);        // branchless mask
        }

        // Online softmax. C layout: VGPR r holds row r+8*half; 16-lane xor
        // reductions (1,2,4,8) stay inside each half-wave group.
        float corr[8];
#pragma unroll
        for (int r = 0; r < 8; ++r) {
            float x = st[0][r];
#pragma unroll
            for (int t = 1; t < 8; ++t) x = fmaxf(x, st[t][r]);
            x = fmaxf(x, __shfl_xor(x, 1));
            x = fmaxf(x, __shfl_xor(x, 2));
            x = fmaxf(x, __shfl_xor(x, 4));
            x = fmaxf(x, __shfl_xor(x, 8));
            const float mnew = fmaxf(mrow[r], x);
            corr[r] = __expf(mrow[r] - mnew);
            mrow[r] = mnew;
        }
#pragma unroll
        for (int t = 0; t < 8; ++t)
#pragma unroll
            for (int r = 0; r < 8; ++r)
                st[t][r] = __expf(st[t][r] - mrow[r]);      // masked cols -> 0
#pragma unroll
        for (int r = 0; r < 8; ++r) {
            float sum = 0.f;
#pragma unroll
            for (int t = 0; t < 8; ++t) sum += st[t][r];
            sum += __shfl_xor(sum, 1);
            sum += __shfl_xor(sum, 2);
            sum += __shfl_xor(sum, 4);
            sum += __shfl_xor(sum, 8);
            lrow[r] = lrow[r] * corr[r] + sum;
        }
#pragma unroll
        for (int t = 0; t < 8; ++t)
#pragma unroll
            for (int r = 0; r < 8; ++r) o[t][r] *= corr[r];

        // All waves done reading current Ks -> recycle it as the P buffer.
        // (TDM is writing the OTHER buffer, so no conflict.)
        __syncthreads();
        __bf16* Ps = Ks + wave * 16 * DIM;      // wave-private 16x128 region
#pragma unroll
        for (int t = 0; t < 8; ++t)
#pragma unroll
            for (int r = 0; r < 8; ++r)
                Ps[(r + 8 * half) * DIM + t * 16 + col] = (__bf16)st[t][r];
        // Same-wave DS ops complete in order; just stop compiler reordering.
        __builtin_amdgcn_wave_barrier();

        bf16x16 pa[4];
        const __bf16* prow = Ps + col * DIM;    // lane = row for A-frags
#pragma unroll
        for (int c = 0; c < 4; ++c) pa[c] = load_a_frag(prow, c, half);

        // O += P V  (B-frag from V^T: lane = out-d col, k contiguous)
#pragma unroll
        for (int t = 0; t < 8; ++t) {
            const __bf16* vr = Vts + (t * 16 + col) * 128;
#pragma unroll
            for (int c = 0; c < 4; ++c) {
                bf16x16 bfrag = *(const bf16x16*)(vr + c * 32 + half * 16);
                o[t] = wmma_bf16(pa[c], bfrag, o[t]);
            }
        }
    }

    // Normalize; zero rows with i >= L (reference zeroes masked rows).
    float* orow = out + ((size_t)b * SEQ + qn) * DIM;
#pragma unroll
    for (int r = 0; r < 8; ++r) {
        const int m = r + 8 * half;
        const bool valid = (qn + m) < L;
        const float inv = (valid && lrow[r] > 0.f) ? (1.f / lrow[r]) : 0.f;
#pragma unroll
        for (int t = 0; t < 8; ++t)
            orow[(size_t)m * DIM + t * 16 + col] = o[t][r] * inv;
    }
}

// ---------------------------------------------------------------------------
extern "C" void kernel_launch(void* const* d_in, const int* in_sizes, int n_in,
                              void* d_out, int out_size, void* d_ws, size_t ws_size,
                              hipStream_t stream) {
    const float* x          = (const float*)d_in[0];
    const int*   valid_lens = (const int*)d_in[1];
    const float* Wq = (const float*)d_in[2];
    const float* bq = (const float*)d_in[3];
    const float* Wk = (const float*)d_in[4];
    const float* bk = (const float*)d_in[5];
    const float* Wv = (const float*)d_in[6];
    const float* bv = (const float*)d_in[7];
    float* out = (float*)d_out;

    __bf16* ws = (__bf16*)d_ws;
    const size_t per = (size_t)BATCH * SEQ * DIM;     // 8M elems = 16MB each
    __bf16* qg  = ws;
    __bf16* kg  = ws + per;
    __bf16* vtg = ws + 2 * per;                       // V stored transposed

    qkv_proj_kernel<<<dim3((BATCH * SEQ) / 128), dim3(256), 0, stream>>>(
        x, Wq, bq, Wk, bk, Wv, bv, qg, kg, vtg);
    attn_kernel<<<dim3(BATCH * 8), dim3(256), 0, stream>>>(
        qg, kg, vtg, valid_lens, out);
}
